// QRNNCell_18502719111534
// MI455X (gfx1250) — matrix operand order repaired and checked by reading, not compiled, not run on hardware
//
#include <hip/hip_runtime.h>
#include <hip/hip_bf16.h>
#include <math.h>

// Problem constants (from reference setup_inputs)
#define L_SEQ   2048
#define BATCH   16
#define N_IN    512
#define N_OUT   512
#define N_COL   (3 * N_OUT)            // 1536
#define M_ROWS  (L_SEQ * BATCH)        // 32768

typedef __attribute__((ext_vector_type(16))) __bf16       bf16x16;
typedef __attribute__((ext_vector_type(8)))  float        f32x8;
typedef __attribute__((ext_vector_type(4)))  float        f32x4;
typedef __attribute__((ext_vector_type(4)))  unsigned int u32x4;

#define LDS_STRIDE 40   // 32 bf16 + 8 pad; 80B rows -> 16B aligned, odd bank stride

// ---------------------------------------------------------------------------
// Kernel 1: u = x @ W  (bf16 WMMA, f32 accumulate), u stored f32 row-major
// grid = (N_COL/128, M_ROWS/128) = (12, 256), block = 256 threads (8 waves)
// Double-buffered LDS: one barrier per K-step, global loads of tile k+1
// overlap the 8 WMMAs of tile k.
// ---------------------------------------------------------------------------
__global__ __launch_bounds__(256)
void qrnn_gemm_bf16_wmma(const float* __restrict__ x,   // (32768, 512)
                         const float* __restrict__ w,   // (512, 1536)
                         float* __restrict__ u)         // (32768, 1536)
{
    __shared__ __bf16 Alds[2][128 * LDS_STRIDE];   // [m][k]
    __shared__ __bf16 Blds[2][128 * LDS_STRIDE];   // [n][k] (transposed W)

    const int tid  = threadIdx.x;
    const int lane = tid & 31;
    const int wave = tid >> 5;
    const int wm   = wave >> 2;                 // 0..1  (M direction)
    const int wn   = wave & 3;                  // 0..3  (N direction)
    const int rowBase = blockIdx.y * 128;
    const int colBase = blockIdx.x * 128;

    // ---- per-thread staging coordinates (loop invariant) ----
    int ar[4], ak[4];                 // A: float4 slots, k-fast
#pragma unroll
    for (int i = 0; i < 4; ++i) {
        const int idx = tid + i * 256;          // 0..1023
        ar[i] = idx >> 3;                       // row 0..127
        ak[i] = (idx & 7) << 2;                 // k offset 0..28
    }
    int bk[2], bn[2];                 // B: 2(k) x 4(n) micro-tiles
#pragma unroll
    for (int i = 0; i < 2; ++i) {
        const int p = tid + i * 256;            // 0..511 micro-tiles
        bk[i] = (p >> 5) << 1;                  // even k: 0..30
        bn[i] = (p & 31) << 2;                  // n offset 0..124
    }

    f32x8 acc[4][2];
#pragma unroll
    for (int mi = 0; mi < 4; ++mi)
#pragma unroll
        for (int ni = 0; ni < 2; ++ni)
#pragma unroll
            for (int r = 0; r < 8; ++r)
                acc[mi][ni][r] = 0.0f;

    union Frag { bf16x16 v; u32x4 q[2]; };

    const int laneLow = lane & 15;
    // A fragment K chunks (elements): lane<16 -> {0..7,16..23}; lane>=16 -> {8..15,24..31}
    const int kA0 = (lane & 16) ? 8 : 0;
    // B fragment K chunks (elements): lane<16 -> {0..7,16..23}; lane>=16 -> {8..15,24..31}
    const int kB0 = (lane & 16) ? 8 : 0;

    // ---- staging helpers ----
    f32x4 aReg[4];
    f32x4 bReg[2][2];

    auto loadTile = [&](int k0) {
#pragma unroll
        for (int i = 0; i < 4; ++i)
            aReg[i] = *(const f32x4*)(x + (size_t)(rowBase + ar[i]) * N_IN + k0 + ak[i]);
#pragma unroll
        for (int i = 0; i < 2; ++i) {
            bReg[i][0] = *(const f32x4*)(w + (size_t)(k0 + bk[i] + 0) * N_COL + colBase + bn[i]);
            bReg[i][1] = *(const f32x4*)(w + (size_t)(k0 + bk[i] + 1) * N_COL + colBase + bn[i]);
        }
    };

    auto storeTile = [&](int buf) {
#pragma unroll
        for (int i = 0; i < 4; ++i) {           // 4 contiguous bf16 -> b64 store
            __bf16* d = &Alds[buf][ar[i] * LDS_STRIDE + ak[i]];
            d[0] = (__bf16)aReg[i][0]; d[1] = (__bf16)aReg[i][1];
            d[2] = (__bf16)aReg[i][2]; d[3] = (__bf16)aReg[i][3];
        }
#pragma unroll
        for (int i = 0; i < 2; ++i)
#pragma unroll
            for (int c = 0; c < 4; ++c) {       // k-pair contiguous -> b32 store
                __bf16* d = &Blds[buf][(bn[i] + c) * LDS_STRIDE + bk[i]];
                d[0] = (__bf16)bReg[i][0][c];
                d[1] = (__bf16)bReg[i][1][c];
            }
    };

    auto compute = [&](int buf) {
        Frag a[4], b[2];
#pragma unroll
        for (int mi = 0; mi < 4; ++mi) {
            const __bf16* p = &Alds[buf][(wm * 64 + mi * 16 + laneLow) * LDS_STRIDE];
            a[mi].q[0] = *(const u32x4*)(p + kA0);
            a[mi].q[1] = *(const u32x4*)(p + kA0 + 16);
        }
#pragma unroll
        for (int ni = 0; ni < 2; ++ni) {
            const __bf16* p = &Blds[buf][(wn * 32 + ni * 16 + laneLow) * LDS_STRIDE];
            b[ni].q[0] = *(const u32x4*)(p + kB0);
            b[ni].q[1] = *(const u32x4*)(p + kB0 + 16);
        }
#pragma unroll
        for (int mi = 0; mi < 4; ++mi)
#pragma unroll
            for (int ni = 0; ni < 2; ++ni)
                acc[mi][ni] = __builtin_amdgcn_wmma_f32_16x16x32_bf16(
                    false, a[mi].v, false, b[ni].v,
                    (short)0, acc[mi][ni], false, false);
    };

    // ---- pipelined main loop: 16 K-steps of 32 ----
    loadTile(0);
    storeTile(0);
    __syncthreads();
#pragma unroll
    for (int it = 0; it < 16; ++it) {
        const int cur = it & 1;
        if (it + 1 < 16) loadTile((it + 1) * 32);   // global loads overlap WMMAs
        compute(cur);
        if (it + 1 < 16) storeTile(cur ^ 1);        // fill the other buffer
        __syncthreads();                            // one barrier per K-step
    }

    // ---- epilogue: C/D layout -> coalesced f32 stores of u ----
    // VGPR r: lanes 0-15 => M=r, N=lane; lanes 16-31 => M=8+r, N=lane-16
    const int mHalf = (lane & 16) ? 8 : 0;
#pragma unroll
    for (int mi = 0; mi < 4; ++mi) {
#pragma unroll
        for (int ni = 0; ni < 2; ++ni) {
            const int n = colBase + wn * 32 + ni * 16 + laneLow;
#pragma unroll
            for (int r = 0; r < 8; ++r) {
                const int m = rowBase + wm * 64 + mi * 16 + mHalf + r;
                u[(size_t)m * N_COL + n] = acc[mi][ni][r];
            }
        }
    }
}

// ---------------------------------------------------------------------------
// Kernel 2: fused bias + activations + fo-pool scan + output gate
// one thread per (batch, channel): 8192 threads = 32 blocks x 256
// ---------------------------------------------------------------------------
__global__ __launch_bounds__(256)
void qrnn_scan(const float* __restrict__ u,     // (32768, 1536)
               const float* __restrict__ bias,  // (1536,) laid out [3][512]
               const float* __restrict__ c0,    // (16, 512)
               float* __restrict__ h,           // (2048, 16, 512)
               float* __restrict__ cfin)        // (16, 512)
{
    const int ch = blockIdx.x * 256 + threadIdx.x;   // 0..8191
    const int b  = ch >> 9;                          // batch
    const int j  = ch & 511;                         // channel

    const float bz = bias[j];
    const float bf = bias[N_OUT + j];
    const float bo = bias[2 * N_OUT + j];

    float c = c0[ch];
    const float* up = u + (size_t)b * N_COL + (size_t)j * 3;

    for (int t = 0; t < L_SEQ; ++t) {
        const float* row = up + (size_t)t * (BATCH * N_COL);
        const float u0 = row[0];
        const float u1 = row[1];
        const float u2 = row[2];
        const float z = tanhf(u0 + bz);
        const float f = 1.0f / (1.0f + expf(-(u1 + bf)));
        const float o = 1.0f / (1.0f + expf(-(u2 + bo)));
        c = f * c + (1.0f - f) * z;
        h[(size_t)t * (BATCH * N_OUT) + ch] = o * tanhf(c);
    }
    cfin[ch] = c;
}

// ---------------------------------------------------------------------------
extern "C" void kernel_launch(void* const* d_in, const int* in_sizes, int n_in,
                              void* d_out, int out_size, void* d_ws, size_t ws_size,
                              hipStream_t stream)
{
    (void)in_sizes; (void)n_in; (void)out_size; (void)ws_size;

    const float* x    = (const float*)d_in[0];  // (2048,16,512)
    const float* w    = (const float*)d_in[1];  // (512,1536)
    const float* bias = (const float*)d_in[2];  // (1536,)
    const float* c0   = (const float*)d_in[3];  // (16,512)

    float* u    = (float*)d_ws;                          // needs 32768*1536*4 = 192MB
    float* h    = (float*)d_out;                         // 2048*16*512 floats
    float* cfin = h + (size_t)L_SEQ * BATCH * N_OUT;     // then 16*512 floats

    dim3 grid(N_COL / 128, M_ROWS / 128);                // (12, 256)
    qrnn_gemm_bf16_wmma<<<grid, 256, 0, stream>>>(x, w, u);

    qrnn_scan<<<(BATCH * N_OUT) / 256, 256, 0, stream>>>(u, bias, c0, h, cfin);
}
